// BiLSTM_CRF_41223096107594
// MI455X (gfx1250) — matrix operand order repaired and checked by reading, not compiled, not run on hardware
//
#include <hip/hip_runtime.h>
#include <hip/hip_bf16.h>

// ---------------------------------------------------------------------------
// BiLSTM-CRF for MI455X (gfx1250, wave32, WMMA)
//   T=8192, EMBED=256, H=256 (per dir), 4H=1024, N_TAGS=16
// Strategy: one-time f32->bf16 conversion of all GEMM operands, then pure
// load+WMMA inner loops. Serial LSTM keeps w_hh resident in VGPRs as WMMA
// B-fragments and h in LDS as bf16.
// ---------------------------------------------------------------------------

#define T_LEN 8192
#define EMB   256
#define HDIM  256
#define GDIM  1024          // 4*H gates per direction
#define NTAGS 16
#define START_IX 14
#define STOP_IX  15

typedef __attribute__((ext_vector_type(16))) __bf16 v16bf;
typedef __attribute__((ext_vector_type(8)))  __bf16 v8bf;
typedef __attribute__((ext_vector_type(2)))  __bf16 v2bf;
typedef __attribute__((ext_vector_type(8)))  float  v8f;

__device__ __forceinline__ __bf16 to_bf16(float f) {
    unsigned u = __builtin_bit_cast(unsigned, f);
    unsigned r = (u + 0x7FFFu + ((u >> 16) & 1u)) >> 16;   // RNE
    unsigned short s = (unsigned short)r;
    return __builtin_bit_cast(__bf16, s);
}

__device__ __forceinline__ v2bf pk2_bf16(float a, float b) {
#if __has_builtin(__builtin_amdgcn_cvt_pk_bf16_f32)
    auto t = __builtin_amdgcn_cvt_pk_bf16_f32(a, b);
    return __builtin_bit_cast(v2bf, t);
#else
    v2bf r; r[0] = to_bf16(a); r[1] = to_bf16(b); return r;
#endif
}

__device__ __forceinline__ v16bf cat16(v8bf lo, v8bf hi) {
    return __builtin_shufflevector(lo, hi,
        0, 1, 2, 3, 4, 5, 6, 7, 8, 9, 10, 11, 12, 13, 14, 15);
}

__device__ __forceinline__ float sigmoidf_(float x) {
    return 1.0f / (1.0f + expf(-x));
}

// ---------------------------------------------------------------------------
// Prep 0a: bulk f32 -> bf16 conversion (weights). n2 = element pairs.
// ---------------------------------------------------------------------------
__global__ void __launch_bounds__(256)
cvt_bf16_kernel(const float* __restrict__ src, unsigned short* __restrict__ dst,
                int n2)
{
    int i = blockIdx.x * 256 + threadIdx.x;
    if (i < n2) {
        float2 v = ((const float2*)src)[i];
        ((v2bf*)dst)[i] = pk2_bf16(v.x, v.y);
    }
}

// ---------------------------------------------------------------------------
// Prep 0b: embedding gather + bf16 convert. grid = T, block = 64 (4 elem/thr).
// ---------------------------------------------------------------------------
__global__ void __launch_bounds__(64)
embed_gather_kernel(const int* __restrict__ sentence,
                    const float* __restrict__ embed,
                    unsigned short* __restrict__ emb_bf)
{
    const int t = blockIdx.x;
    const float* row = embed + (long)sentence[t] * EMB;
    unsigned short* dst = emb_bf + (long)t * EMB;
    const int i = threadIdx.x;
    float4 v = ((const float4*)row)[i];
    ((v2bf*)dst)[2 * i]     = pk2_bf16(v.x, v.y);
    ((v2bf*)dst)[2 * i + 1] = pk2_bf16(v.z, v.w);
}

// ---------------------------------------------------------------------------
// Kernel 1: input projection (both directions), pure bf16 load + WMMA.
//   xw[t][n] = emb[t] . w_ih[n] + (b_ih[n] + b_hh[n])
// One wave per block; 16(M) x 64(N) tile, K = 256.
// grid = (T/16, 1024/64, 2 dirs), block = 32.
// ---------------------------------------------------------------------------
__global__ void __launch_bounds__(32)
embed_gemm_kernel(const unsigned short* __restrict__ emb_bf,
                  const unsigned short* __restrict__ wihf_bf,
                  const unsigned short* __restrict__ wihb_bf,
                  const float* __restrict__ b_ih_f, const float* __restrict__ b_hh_f,
                  const float* __restrict__ b_ih_b, const float* __restrict__ b_hh_b,
                  float* __restrict__ xw_f, float* __restrict__ xw_b)
{
    const int dir = blockIdx.z;
    const unsigned short* w_ih = dir ? wihb_bf : wihf_bf;
    const float* bi = dir ? b_ih_b : b_ih_f;
    const float* bh = dir ? b_hh_b : b_hh_f;
    float* xw       = dir ? xw_b   : xw_f;

    const int lane  = threadIdx.x;
    const int lg    = lane >> 4;
    const int col   = lane & 15;
    const int tileM = blockIdx.x;
    const int nBase = blockIdx.y * 64;

    const int row = tileM * 16 + col;
    const unsigned short* erow = emb_bf + (long)row * EMB;

    v8f acc[4] = {v8f{}, v8f{}, v8f{}, v8f{}};

    for (int kt = 0; kt < 8; ++kt) {
        // A frag: lanes0-15 K=0..15 ; lanes16-31 K=16..31
        const int k0 = kt * 32 + lg * 16;
        v8bf alo = *(const v8bf*)(erow + k0);
        v8bf ahi = *(const v8bf*)(erow + k0 + 8);
        v16bf a = cat16(alo, ahi);

        #pragma unroll
        for (int nt = 0; nt < 4; ++nt) {
            const int n = nBase + nt * 16 + col;
            const unsigned short* wr = w_ih + (long)n * EMB + kt * 32 + lg * 16;
            v16bf b = cat16(*(const v8bf*)wr, *(const v8bf*)(wr + 8));
            acc[nt] = __builtin_amdgcn_wmma_f32_16x16x32_bf16(
                false, a, false, b, (short)0, acc[nt], false, false);
        }
    }

    #pragma unroll
    for (int nt = 0; nt < 4; ++nt) {
        const int n = nBase + nt * 16 + col;
        const float bias = bi[n] + bh[n];
        #pragma unroll
        for (int j = 0; j < 8; ++j) {
            const int m = tileM * 16 + j + 8 * lg;
            xw[(long)m * GDIM + n] = acc[nt][j] + bias;
        }
    }
}

// ---------------------------------------------------------------------------
// Kernel 2: LSTM recurrence, one workgroup per direction (grid = 2).
// 1024 threads = 32 waves; w_hh lives in VGPRs as bf16 B-fragments
// (2 N-tiles x 8 K-tiles x 8 VGPRs = 128 VGPRs/wave). h kept in LDS as bf16.
// Writes hs directly in bf16 for the feats GEMM.
// ---------------------------------------------------------------------------
__global__ void __launch_bounds__(1024)
lstm_rec_kernel(const unsigned short* __restrict__ whhf_bf,
                const unsigned short* __restrict__ whhb_bf,
                const float* __restrict__ xw_f, const float* __restrict__ xw_b,
                const float* __restrict__ h0,   const float* __restrict__ c0,
                unsigned short* __restrict__ hsf_bf,
                unsigned short* __restrict__ hsb_bf)
{
    const int dir = blockIdx.x;
    const unsigned short* w_hh = dir ? whhb_bf : whhf_bf;
    const float* xw = dir ? xw_b : xw_f;
    unsigned short* hs_bf = dir ? hsb_bf : hsf_bf;

    const int tid  = threadIdx.x;
    const int lane = tid & 31;
    const int wave = tid >> 5;
    const int lg   = lane >> 4;
    const int col  = lane & 15;

    __shared__ __align__(16) unsigned short hbf_s[HDIM];
    __shared__ float z_s[GDIM];

    // Persistent weight fragments.
    v16bf Bf[2][8];
    #pragma unroll
    for (int nt = 0; nt < 2; ++nt) {
        const int n = (wave * 2 + nt) * 16 + col;
        #pragma unroll
        for (int kt = 0; kt < 8; ++kt) {
            const unsigned short* wr = w_hh + (long)n * HDIM + kt * 32 + lg * 16;
            Bf[nt][kt] = cat16(*(const v8bf*)wr, *(const v8bf*)(wr + 8));
        }
    }

    if (tid < HDIM) {
        hbf_s[tid] = __builtin_bit_cast(unsigned short, to_bf16(h0[dir * HDIM + tid]));
    }
    float c = (tid < HDIM) ? c0[dir * HDIM + tid] : 0.0f;
    __syncthreads();

    const int n0 = (wave * 2 + 0) * 16 + lane;   // valid when lane < 16
    const int n1 = (wave * 2 + 1) * 16 + lane;

    for (int s = 0; s < T_LEN; ++s) {
        const int t = dir ? (T_LEN - 1 - s) : s;

        // load xw early so it overlaps the WMMA chain
        float x0 = 0.0f, x1 = 0.0f;
        if (lane < 16) {
            const long base = (long)t * GDIM;
            x0 = xw[base + n0];
            x1 = xw[base + n1];
        }

        v8f acc0 = v8f{}, acc1 = v8f{};
        #pragma unroll
        for (int kt = 0; kt < 8; ++kt) {
            const int k0 = kt * 32 + lg * 16;
            v8bf lo = *(const v8bf*)(hbf_s + k0);        // ds_load_b128
            v8bf hi = *(const v8bf*)(hbf_s + k0 + 8);
            v16bf a;
            if (col == 0) a = cat16(lo, hi);             // only row 0 of A live
            else          a = v16bf{};
            acc0 = __builtin_amdgcn_wmma_f32_16x16x32_bf16(
                false, a, false, Bf[0][kt], (short)0, acc0, false, false);
            acc1 = __builtin_amdgcn_wmma_f32_16x16x32_bf16(
                false, a, false, Bf[1][kt], (short)0, acc1, false, false);
        }

        if (lane < 16) {
            z_s[n0] = acc0[0] + x0;    // D row 0: VGPR0, lanes 0..15, N=lane
            z_s[n1] = acc1[0] + x1;
        }
        if (s + 1 < T_LEN) {
            const int tn = dir ? (T_LEN - 2 - s) : (s + 1);
            __builtin_prefetch(xw + (long)tn * GDIM + tid, 0, 1);
        }
        __syncthreads();

        if (tid < HDIM) {
            const float zi = z_s[tid];
            const float zf = z_s[HDIM + tid];
            const float zg = z_s[2 * HDIM + tid];
            const float zo = z_s[3 * HDIM + tid];
            const float ig = sigmoidf_(zi);
            const float fg = sigmoidf_(zf);
            const float gg = tanhf(zg);
            const float og = sigmoidf_(zo);
            c = fg * c + ig * gg;
            const float h = og * tanhf(c);
            const unsigned short hb = __builtin_bit_cast(unsigned short, to_bf16(h));
            hbf_s[tid] = hb;
            hs_bf[(long)t * HDIM + tid] = hb;
        }
        __syncthreads();
    }
}

// ---------------------------------------------------------------------------
// Kernel 3: feats[t][tag] = [hs_f|hs_b][t] . W_out[tag] + b_out[tag]
// One wave per 16 rows; K = 512 -> 16 WMMAs, all-bf16 loads.
// ---------------------------------------------------------------------------
__global__ void __launch_bounds__(32)
feats_kernel(const unsigned short* __restrict__ hsf_bf,
             const unsigned short* __restrict__ hsb_bf,
             const unsigned short* __restrict__ wout_bf,
             const float* __restrict__ b_out,
             float* __restrict__ feats)
{
    const int lane = threadIdx.x;
    const int lg   = lane >> 4;
    const int col  = lane & 15;
    const int tile = blockIdx.x;
    const int row  = tile * 16 + col;

    v8f acc = v8f{};
    for (int kt = 0; kt < 16; ++kt) {
        const unsigned short* src = (kt < 8)
            ? (hsf_bf + (long)row * HDIM + kt * 32)
            : (hsb_bf + (long)row * HDIM + (kt - 8) * 32);
        const int k0 = lg * 16;
        v16bf a = cat16(*(const v8bf*)(src + k0), *(const v8bf*)(src + k0 + 8));

        const unsigned short* wr = wout_bf + (long)col * 512 + kt * 32 + lg * 16;
        v16bf b = cat16(*(const v8bf*)wr, *(const v8bf*)(wr + 8));

        acc = __builtin_amdgcn_wmma_f32_16x16x32_bf16(
            false, a, false, b, (short)0, acc, false, false);
    }

    const float bias = b_out[col];
    #pragma unroll
    for (int j = 0; j < 8; ++j) {
        const int t = tile * 16 + j + 8 * lg;
        feats[(long)t * NTAGS + col] = acc[j] + bias;
    }
}

// ---------------------------------------------------------------------------
// Kernel 4: Viterbi forward pass + backtrack. Single wave.
// ---------------------------------------------------------------------------
__global__ void __launch_bounds__(32)
viterbi_kernel(const float* __restrict__ feats,
               const float* __restrict__ transition,
               float* __restrict__ out,
               int* __restrict__ bps)
{
    const int lane = threadIdx.x;
    __shared__ float fv_s[NTAGS];

    float tr[NTAGS];
    if (lane < NTAGS) {
        #pragma unroll
        for (int j = 0; j < NTAGS; ++j) tr[j] = transition[lane * NTAGS + j];
        fv_s[lane] = (lane == START_IX) ? 0.0f : -10000.0f;
    }
    __syncthreads();

    for (int t = 0; t < T_LEN; ++t) {
        float best = -3.4e38f;
        int   arg  = 0;
        if (lane < NTAGS) {
            #pragma unroll
            for (int j = 0; j < NTAGS; ++j) {
                const float v = fv_s[j] + tr[j];
                if (v > best) { best = v; arg = j; }
            }
            best += feats[t * NTAGS + lane];
            bps[t * NTAGS + lane] = arg;
        }
        __syncthreads();
        if (lane < NTAGS) fv_s[lane] = best;
        __syncthreads();
    }

    if (lane == 0) {
        float bb = -3.4e38f;
        int id_last = 0;
        for (int i = 0; i < NTAGS; ++i) {
            const float v = fv_s[i] + transition[i * NTAGS + STOP_IX];
            if (v > bb) { bb = v; id_last = i; }
        }
        int idc = id_last;
        for (int t = T_LEN - 1; t >= 0; --t) {
            out[t] = (float)idc;
            idc = bps[t * NTAGS + idc];
        }
    }
}

// ---------------------------------------------------------------------------
// Launch
// ---------------------------------------------------------------------------
extern "C" void kernel_launch(void* const* d_in, const int* in_sizes, int n_in,
                              void* d_out, int out_size, void* d_ws, size_t ws_size,
                              hipStream_t stream) {
    const int*   sentence   = (const int*)  d_in[0];
    const float* embed      = (const float*)d_in[1];
    const float* w_ih_f     = (const float*)d_in[2];
    const float* w_hh_f     = (const float*)d_in[3];
    const float* b_ih_f     = (const float*)d_in[4];
    const float* b_hh_f     = (const float*)d_in[5];
    const float* w_ih_b     = (const float*)d_in[6];
    const float* w_hh_b     = (const float*)d_in[7];
    const float* b_ih_b     = (const float*)d_in[8];
    const float* b_hh_b     = (const float*)d_in[9];
    const float* W_out      = (const float*)d_in[10];
    const float* b_out      = (const float*)d_in[11];
    const float* transition = (const float*)d_in[12];
    const float* h0         = (const float*)d_in[13];
    const float* c0         = (const float*)d_in[14];

    // ---- workspace layout (16B-aligned segments) ----
    float* ws    = (float*)d_ws;
    float* xw_f  = ws;                                  // T*1024 f32
    float* xw_b  = xw_f + (long)T_LEN * GDIM;           // T*1024 f32
    float* feats = xw_b + (long)T_LEN * GDIM;           // T*16  f32
    int*   bps   = (int*)(feats + (long)T_LEN * NTAGS); // T*16  i32
    unsigned short* emb_bf  = (unsigned short*)(bps + (long)T_LEN * NTAGS); // T*256
    unsigned short* wihf_bf = emb_bf  + (long)T_LEN * EMB;   // 1024*256
    unsigned short* wihb_bf = wihf_bf + GDIM * EMB;
    unsigned short* whhf_bf = wihb_bf + GDIM * EMB;
    unsigned short* whhb_bf = whhf_bf + GDIM * HDIM;
    unsigned short* wout_bf = whhb_bf + GDIM * HDIM;         // 16*512
    unsigned short* hsf_bf  = wout_bf + NTAGS * 512;         // T*256
    unsigned short* hsb_bf  = hsf_bf  + (long)T_LEN * HDIM;  // T*256

    // ---- prep: one-time bf16 conversions ----
    const int wPairs = (GDIM * EMB) / 2;   // 131072
    cvt_bf16_kernel<<<wPairs / 256, 256, 0, stream>>>(w_ih_f, wihf_bf, wPairs);
    cvt_bf16_kernel<<<wPairs / 256, 256, 0, stream>>>(w_ih_b, wihb_bf, wPairs);
    cvt_bf16_kernel<<<wPairs / 256, 256, 0, stream>>>(w_hh_f, whhf_bf, wPairs);
    cvt_bf16_kernel<<<wPairs / 256, 256, 0, stream>>>(w_hh_b, whhb_bf, wPairs);
    const int oPairs = (NTAGS * 512) / 2;  // 4096
    cvt_bf16_kernel<<<oPairs / 256, 256, 0, stream>>>(W_out, wout_bf, oPairs);
    embed_gather_kernel<<<T_LEN, 64, 0, stream>>>(sentence, embed, emb_bf);

    // ---- main pipeline ----
    dim3 g1(T_LEN / 16, GDIM / 64, 2);
    embed_gemm_kernel<<<g1, 32, 0, stream>>>(
        emb_bf, wihf_bf, wihb_bf,
        b_ih_f, b_hh_f, b_ih_b, b_hh_b, xw_f, xw_b);

    lstm_rec_kernel<<<2, 1024, 0, stream>>>(
        whhf_bf, whhb_bf, xw_f, xw_b, h0, c0, hsf_bf, hsb_bf);

    feats_kernel<<<T_LEN / 16, 32, 0, stream>>>(hsf_bf, hsb_bf, wout_bf, b_out, feats);

    viterbi_kernel<<<1, 32, 0, stream>>>(feats, transition, (float*)d_out, bps);
}
